// RecursiveCompressor_45784351375667
// MI455X (gfx1250) — matrix operand order repaired and checked by reading, not compiled, not run on hardware
//
#include <hip/hip_runtime.h>
#include <stdint.h>

#define D_MODEL 512
#define NHEADS  8
#define HDIM    64
#define D_FF    2048
#define CHUNK   64
#define COMP    8
#define BATCH   4
#define SEQ     4096

typedef __attribute__((ext_vector_type(16))) __bf16 v16bf;
typedef __attribute__((ext_vector_type(8)))  float  v8f;
typedef __attribute__((ext_vector_type(4)))  unsigned int v4u;
typedef __attribute__((ext_vector_type(8)))  int v8i;
typedef __attribute__((ext_vector_type(4)))  int v4i;

union BFrag { v16bf v; uint4 u[2]; };

__device__ __forceinline__ unsigned short f2bf(float f) {
  unsigned int u = __float_as_uint(f);
  u += 0x7FFFu + ((u >> 16) & 1u);   // round-to-nearest-even
  return (unsigned short)(u >> 16);
}

// ---------------------------------------------------------------------------
// TDM: issue a tensor_load_to_lds for a 64x64 bf16 tile.
//   ldsOff    : LDS byte offset of destination (tile stored row-major 64x64)
//   gbase     : global address of tile start (element [row0, col0])
//   rowStride : elements per tensor row (tensor_dim0 == tensor_dim0_stride)
//   rows      : total tensor rows (tensor_dim1)
// D# layout per CDNA5 ISA ch.8 (group0 128b, group1 256b; groups 2/3 zero for 2D).
// ---------------------------------------------------------------------------
__device__ __forceinline__ void tdm_load_tile64x64(
    unsigned ldsOff, const unsigned short* gbase, int rowStride, int rows)
{
  const unsigned long long ga = (unsigned long long)(uintptr_t)gbase;
  v4u g0;
  g0.x = 1u;                                   // count=1 (valid user descriptor)
  g0.y = ldsOff;                               // lds_addr (bytes)
  g0.z = (unsigned)ga;                         // global_addr[31:0]
  g0.w = (unsigned)(ga >> 32) | (2u << 30);    // global_addr[56:32] | type=2
  const unsigned td0 = (unsigned)rowStride;    // tensor_dim0 (data_size units)
  const unsigned td1 = (unsigned)rows;         // tensor_dim1
  const unsigned long long s0 = (unsigned long long)rowStride; // dim0 stride
  v8i g1;
  g1[0] = (int)(1u << 16);                             // data_size=1 (2 bytes)
  g1[1] = (int)((td0 & 0xFFFFu) << 16);                // tensor_dim0[15:0]
  g1[2] = (int)((td0 >> 16) | ((td1 & 0xFFFFu) << 16));// dim0[31:16] | dim1[15:0]
  g1[3] = (int)((td1 >> 16) | (64u << 16));            // dim1[31:16] | tile_dim0=64
  g1[4] = (int)64u;                                    // tile_dim1=64, tile_dim2=0
  g1[5] = (int)(unsigned)(s0 & 0xFFFFFFFFu);           // dim0_stride[31:0]
  g1[6] = (int)(unsigned)(s0 >> 32);                   // dim0_stride[47:32]
  g1[7] = 0;
  const v4i z4 = {0, 0, 0, 0};
#if __has_include(<hip/amd_detail/amd_gfx1250_TDM.h>)
  const v8i z8 = {0, 0, 0, 0, 0, 0, 0, 0};
  __builtin_amdgcn_tensor_load_to_lds(g0, g1, z4, z4, z8, 0);
#else
  __builtin_amdgcn_tensor_load_to_lds(g0, g1, z4, z4, 0);
#endif
}

// ---------------------------------------------------------------------------
// GEMM: C[M,N] = A[M,K](bf16) @ W[N,K](bf16)^T (+bias[N]) (+res[M,N])
// Block tile 64x64, K-stage 64, TDM double-buffered LDS staging.
// Wave 0 issues 2 TDM tile loads per stage (TENSORcnt); all 8 waves compute
// 16(M)x32(N) slabs from the previous LDS buffer.
// A-frag (ISA 16-bit A 16x32): lane L -> row M=L&15, K = (L>>4)*8 + {0..7,16..23}
// B-frag (ISA 16-bit B 32x16): lane L -> col N=L&15, K = (L>>4)*8 + {0..7,16..23}
// C/D  : VGPR r, lane L -> M = r + 8*(L>>4), N = L&15
// ---------------------------------------------------------------------------
__global__ __launch_bounds__(256) void gemm_bf16(
    const unsigned short* __restrict__ A, const unsigned short* __restrict__ W,
    const float* __restrict__ bias, const float* res,
    float* C, int M, int N, int K)
{
  __shared__ unsigned short As[2][64][64];
  __shared__ unsigned short Bs[2][64][64];

  const int wave = threadIdx.x >> 5;
  const int lane = threadIdx.x & 31;
  const int waveId = __builtin_amdgcn_readfirstlane(wave); // scalar branch below
  const int m0 = blockIdx.y * 64;
  const int n0 = blockIdx.x * 64;
  const int wm = (wave >> 1) * 16;
  const int wn = (wave & 1) * 32;
  const int half = lane >> 4;
  const int lr = lane & 15;

  const unsigned short* Abase = A + (size_t)m0 * K;
  const unsigned short* Bbase = W + (size_t)n0 * K;
  const unsigned offA0 = (unsigned)(uintptr_t)&As[0][0][0];
  const unsigned offA1 = (unsigned)(uintptr_t)&As[1][0][0];
  const unsigned offB0 = (unsigned)(uintptr_t)&Bs[0][0][0];
  const unsigned offB1 = (unsigned)(uintptr_t)&Bs[1][0][0];

  const int nk = K >> 6;                       // 64-wide K stages

  if (waveId == 0) {
    tdm_load_tile64x64(offA0, Abase, K, M);
    tdm_load_tile64x64(offB0, Bbase, K, N);
  }

  v8f acc0 = {}; v8f acc1 = {};
  for (int s = 0; s < nk; ++s) {
    const int buf = s & 1;
    if (waveId == 0) {
      if (s + 1 < nk) {
        // prefetch next stage into the other buffer (freed by last barrier)
        tdm_load_tile64x64(buf ? offA0 : offA1, Abase + (size_t)(s + 1) * 64, K, M);
        tdm_load_tile64x64(buf ? offB0 : offB1, Bbase + (size_t)(s + 1) * 64, K, N);
        __builtin_amdgcn_s_wait_tensorcnt(2);  // in-order: current buffer done
      } else {
        __builtin_amdgcn_s_wait_tensorcnt(0);
      }
    }
    __syncthreads();                           // LDS stage visible to all waves
#pragma unroll
    for (int kk = 0; kk < 64; kk += 32) {
      BFrag a, b0, b1;
      a.u[0]  = *(const uint4*)(&As[buf][wm + lr][kk + half * 8]);
      a.u[1]  = *(const uint4*)(&As[buf][wm + lr][kk + half * 8 + 16]);
      b0.u[0] = *(const uint4*)(&Bs[buf][wn + lr][kk + half * 8]);
      b0.u[1] = *(const uint4*)(&Bs[buf][wn + lr][kk + half * 8 + 16]);
      b1.u[0] = *(const uint4*)(&Bs[buf][wn + 16 + lr][kk + half * 8]);
      b1.u[1] = *(const uint4*)(&Bs[buf][wn + 16 + lr][kk + half * 8 + 16]);
      acc0 = __builtin_amdgcn_wmma_f32_16x16x32_bf16(false, a.v, false, b0.v,
                                                     (short)0, acc0, false, false);
      acc1 = __builtin_amdgcn_wmma_f32_16x16x32_bf16(false, a.v, false, b1.v,
                                                     (short)0, acc1, false, false);
    }
    __syncthreads();                           // protect buffer being refilled
  }

  const int mb = m0 + wm + half * 8;
  const int c0 = n0 + wn + lr, c1 = n0 + wn + 16 + lr;
  const float bia0 = bias ? bias[c0] : 0.f;
  const float bia1 = bias ? bias[c1] : 0.f;
#pragma unroll
  for (int r = 0; r < 8; ++r) {
    const size_t i0 = (size_t)(mb + r) * N + c0;
    const size_t i1 = (size_t)(mb + r) * N + c1;
    float v0 = acc0[r] + bia0;
    float v1 = acc1[r] + bia1;
    if (res) { v0 += res[i0]; v1 += res[i1]; }
    C[i0] = v0; C[i1] = v1;
  }
}

// ---------------------------------------------------------------------------
// Attention for one (chunk, head): O = softmax(Q K^T / 8 [+mask]) V, all WMMA.
// Q/K/V are f32 [chunk*L + m, D_MODEL] with head slice at column head*64.
// Lq in {8,64}; Lk in {8,64}; LKP = Lk padded to a multiple of 32 for WMMA-K.
// ---------------------------------------------------------------------------
__global__ __launch_bounds__(256) void attn_kernel(
    const float* __restrict__ Q, const float* __restrict__ K,
    const float* __restrict__ V, unsigned short* __restrict__ O,
    int Lq, int Lk, int LKP, int causal, int qBroadcast)
{
  __shared__ unsigned short Qs[64][64];
  __shared__ unsigned short Ks[64][64];
  __shared__ unsigned short Vt[64][64];   // transposed: Vt[d][k]
  __shared__ float          Sc[64][64];
  __shared__ unsigned short Wb[64][64];

  const int chunk = blockIdx.x, head = blockIdx.y;
  const int tid = threadIdx.x;
  const size_t qbase = (qBroadcast ? (size_t)0 : (size_t)chunk * Lq) * D_MODEL + head * HDIM;
  const size_t kbase = (size_t)chunk * Lk * D_MODEL + head * HDIM;

  for (int i = tid; i < 64 * 64; i += 256) {
    const int r = i >> 6, c = i & 63;
    const float q = (r < Lq) ? Q[qbase + (size_t)r * D_MODEL + c] : 0.f;
    const float k = (r < Lk) ? K[kbase + (size_t)r * D_MODEL + c] : 0.f;
    const float v = (r < Lk) ? V[kbase + (size_t)r * D_MODEL + c] : 0.f;
    Qs[r][c] = f2bf(q);
    Ks[r][c] = f2bf(k);
    Vt[c][r] = f2bf(v);
    Wb[r][c] = 0;
  }
  __syncthreads();

  const int wave = tid >> 5, lane = tid & 31;
  const int half = lane >> 4, lr = lane & 15;
  const int Mtiles = (Lq + 15) >> 4;
  const int Ntiles = LKP >> 4;

  // scores = Q @ K^T (K-dim = HDIM = 64)
  for (int t = wave; t < Mtiles * Ntiles; t += 8) {
    const int m0 = (t / Ntiles) * 16, n0 = (t % Ntiles) * 16;
    v8f acc = {};
    for (int k0 = 0; k0 < HDIM; k0 += 32) {
      BFrag a, b;
      a.u[0] = *(const uint4*)(&Qs[m0 + lr][k0 + half * 8]);
      a.u[1] = *(const uint4*)(&Qs[m0 + lr][k0 + half * 8 + 16]);
      b.u[0] = *(const uint4*)(&Ks[n0 + lr][k0 + half * 8]);
      b.u[1] = *(const uint4*)(&Ks[n0 + lr][k0 + half * 8 + 16]);
      acc = __builtin_amdgcn_wmma_f32_16x16x32_bf16(false, a.v, false, b.v,
                                                    (short)0, acc, false, false);
    }
    const int mb = m0 + half * 8, col = n0 + lr;
#pragma unroll
    for (int r = 0; r < 8; ++r) {
      const int row = mb + r;
      float s = acc[r] * 0.125f;             // 1/sqrt(HDIM)
      if (col >= Lk || (causal && col > row)) s = -3.0e38f;
      Sc[row][col] = s;
    }
  }
  __syncthreads();

  // row-wise softmax -> bf16 weights
  if (tid < Lq) {
    float mx = -3.0e38f;
    for (int c = 0; c < LKP; ++c) mx = fmaxf(mx, Sc[tid][c]);
    float sum = 0.f;
    for (int c = 0; c < LKP; ++c) { const float e = __expf(Sc[tid][c] - mx); Sc[tid][c] = e; sum += e; }
    const float inv = 1.f / sum;
    for (int c = 0; c < LKP; ++c) Wb[tid][c] = f2bf(Sc[tid][c] * inv);
  }
  __syncthreads();

  // out = W @ V (K-dim = LKP); B-frag reads Vt[d][k] contiguously
  const size_t obase = (size_t)chunk * Lq * D_MODEL + head * HDIM;
  for (int t = wave; t < Mtiles * 4; t += 8) {
    const int m0 = (t >> 2) * 16, n0 = (t & 3) * 16;
    v8f acc = {};
    for (int k0 = 0; k0 < LKP; k0 += 32) {
      BFrag a, b;
      a.u[0] = *(const uint4*)(&Wb[m0 + lr][k0 + half * 8]);
      a.u[1] = *(const uint4*)(&Wb[m0 + lr][k0 + half * 8 + 16]);
      b.u[0] = *(const uint4*)(&Vt[n0 + lr][k0 + half * 8]);
      b.u[1] = *(const uint4*)(&Vt[n0 + lr][k0 + half * 8 + 16]);
      acc = __builtin_amdgcn_wmma_f32_16x16x32_bf16(false, a.v, false, b.v,
                                                    (short)0, acc, false, false);
    }
    const int mb = m0 + half * 8, col = n0 + lr;
#pragma unroll
    for (int r = 0; r < 8; ++r) {
      const int row = mb + r;
      if (row < Lq) O[obase + (size_t)row * D_MODEL + col] = f2bf(acc[r]);
    }
  }
}

// ---------------------------------------------------------------------------
// LayerNorm: one wave per token (512 elems, 16/lane), emits bf16.
// ---------------------------------------------------------------------------
__global__ __launch_bounds__(256) void ln_kernel(
    const float* __restrict__ x, const float* __restrict__ g,
    const float* __restrict__ b, unsigned short* __restrict__ out, int ntok)
{
  const int tok = blockIdx.x * 8 + (threadIdx.x >> 5);
  if (tok >= ntok) return;
  const int lane = threadIdx.x & 31;
  const float* xr = x + (size_t)tok * D_MODEL;
  float v[16]; float s = 0.f;
#pragma unroll
  for (int i = 0; i < 16; ++i) { v[i] = xr[lane + i * 32]; s += v[i]; }
#pragma unroll
  for (int o = 16; o > 0; o >>= 1) s += __shfl_xor(s, o, 32);
  const float mean = s * (1.f / D_MODEL);
  float var = 0.f;
#pragma unroll
  for (int i = 0; i < 16; ++i) { const float d = v[i] - mean; var += d * d; }
#pragma unroll
  for (int o = 16; o > 0; o >>= 1) var += __shfl_xor(var, o, 32);
  const float rstd = rsqrtf(var * (1.f / D_MODEL) + 1e-5f);
  unsigned short* orow = out + (size_t)tok * D_MODEL;
#pragma unroll
  for (int i = 0; i < 16; ++i) {
    const int c = lane + i * 32;
    orow[c] = f2bf((v[i] - mean) * rstd * g[c] + b[c]);
  }
}

// SiLU-GLU: hh[ntok, 2*D_FF] -> out[ntok, D_FF] bf16
__global__ void siglu_kernel(const float* __restrict__ hh,
                             unsigned short* __restrict__ out, size_t n)
{
  const size_t i = (size_t)blockIdx.x * blockDim.x + threadIdx.x;
  if (i >= n) return;
  const size_t tok = i / D_FF;
  const int c = (int)(i - tok * D_FF);
  const float a = hh[tok * (2 * D_FF) + c];
  const float g = hh[tok * (2 * D_FF) + D_FF + c];
  out[i] = f2bf((a / (1.f + __expf(-a))) * g);
}

__global__ void cvt_bf16_kernel(const float* __restrict__ s,
                                unsigned short* __restrict__ d, size_t n)
{
  const size_t i = (size_t)blockIdx.x * blockDim.x + threadIdx.x;
  if (i < n) d[i] = f2bf(s[i]);
}

__global__ void copy_f32_kernel(const float* __restrict__ s, float* __restrict__ d, size_t n)
{
  const size_t i = (size_t)blockIdx.x * blockDim.x + threadIdx.x;
  if (i < n) d[i] = s[i];
}

// cq (8x512) -> bf16 padded to 64x512 (rows >= 8 zero)
__global__ void padcq_kernel(const float* __restrict__ cq, unsigned short* __restrict__ d)
{
  const int i = blockIdx.x * 256 + threadIdx.x;   // 64*512 total
  if (i >= 64 * D_MODEL) return;
  d[i] = (i < COMP * D_MODEL) ? f2bf(cq[i]) : (unsigned short)0;
}

// forward: dst[(b*COMP+q)*nf+cf] = src[(b*nf+cf)*COMP+q]; inverse swaps roles.
__global__ void permute_kernel(const float* __restrict__ src, float* __restrict__ dst,
                               int B, int nf, int inverse)
{
  const size_t n = (size_t)B * nf * COMP * D_MODEL;
  const size_t i = (size_t)blockIdx.x * blockDim.x + threadIdx.x;
  if (i >= n) return;
  const int d = (int)(i & (D_MODEL - 1));
  const size_t row = i >> 9;
  size_t srow;
  if (!inverse) {
    const int cf = (int)(row % nf);
    const size_t bq = row / nf;
    srow = ((bq >> 3) * nf + cf) * COMP + (bq & 7);
  } else {
    const int q = (int)(row & 7);
    const size_t bcf = row >> 3;
    const int cf = (int)(bcf % nf);
    srow = ((bcf / nf) * COMP + q) * nf + cf;
  }
  dst[i] = src[srow * D_MODEL + d];
}

// outer[(b*nf+cf)*COMP+q] = (cf==0) ? cq[q] : comp_r[(b*nf+cf-1)*COMP+q]
__global__ void build_outer_kernel(const float* __restrict__ comp_r,
                                   const float* __restrict__ cq,
                                   float* __restrict__ outer, int B, int nf)
{
  const size_t n = (size_t)B * nf * COMP * D_MODEL;
  const size_t i = (size_t)blockIdx.x * blockDim.x + threadIdx.x;
  if (i >= n) return;
  const int d = (int)(i & (D_MODEL - 1));
  const size_t row = i >> 9;
  const int q = (int)(row & 7);
  const int cf = (int)((row >> 3) % nf);
  outer[i] = (cf == 0) ? cq[q * D_MODEL + d]
                       : comp_r[(row - COMP) * D_MODEL + d];
}

// (nseq,1,512) -> padded chunks (nseq*64,512): position 0 real, rest zero
__global__ void pad_seq1_kernel(const float* __restrict__ src, float* __restrict__ dst, int nseq)
{
  const size_t n = (size_t)nseq * CHUNK * D_MODEL;
  const size_t i = (size_t)blockIdx.x * blockDim.x + threadIdx.x;
  if (i >= n) return;
  const int p = (int)((i >> 9) & (CHUNK - 1));
  const size_t c = i >> 15;                      // 9 (D) + 6 (CHUNK)
  dst[i] = (p == 0) ? src[c * D_MODEL + (i & (D_MODEL - 1))] : 0.f;
}

__global__ void extract_seq1_kernel(const float* __restrict__ src, float* __restrict__ dst, int nseq)
{
  const size_t n = (size_t)nseq * D_MODEL;
  const size_t i = (size_t)blockIdx.x * blockDim.x + threadIdx.x;
  if (i >= n) return;
  const size_t c = i >> 9;
  dst[i] = src[c * CHUNK * D_MODEL + (i & (D_MODEL - 1))];
}

// ---------------------------------------------------------------------------
extern "C" void kernel_launch(void* const* d_in, const int* in_sizes, int n_in,
                              void* d_out, int out_size, void* d_ws, size_t ws_size,
                              hipStream_t stream)
{
  (void)n_in; (void)out_size; (void)ws_size;

  struct LinP { const float* W; const float* b; unsigned short* Wb; int n, k; };
  struct Mha  { LinP q, k, v, o; };
  struct Ffn  { LinP w1, w2; };
  struct Ln   { const float* g; const float* b; };

  auto in = [&](int i) { return (const float*)d_in[i]; };

  const float *x, *cq;
  Ln ln_enc_mha, ln_enc_ffn, ln_dec_mha, ln_dec_ffn, ln_comp, ln_decomp;
  Mha mha_enc, mha_dec, mha_comp, mha_decomp;
  Ffn ffn_enc, ffn_dec;

  int idx = 0;
  auto lin = [&](LinP& L, int n, int k) { L.W = in(idx++); L.b = in(idx++); L.n = n; L.k = k; };

  if (in_sizes[0] == BATCH * SEQ * D_MODEL) {
    // insertion-order flattening: x first, nested dicts in insertion order
    x = in(idx++); cq = in(idx++);
    auto lnp = [&](Ln& L) { L.g = in(idx++); L.b = in(idx++); };
    lnp(ln_enc_mha); lnp(ln_enc_ffn); lnp(ln_dec_mha); lnp(ln_dec_ffn); lnp(ln_comp); lnp(ln_decomp);
    auto mh = [&](Mha& M) { lin(M.q,512,512); lin(M.k,512,512); lin(M.v,512,512); lin(M.o,512,512); };
    mh(mha_enc); mh(mha_dec); mh(mha_comp); mh(mha_decomp);
    auto ff = [&](Ffn& F) { lin(F.w1, 2 * D_FF, D_MODEL); lin(F.w2, D_MODEL, D_FF); };
    ff(ffn_enc); ff(ffn_dec);
  } else {
    // jax-pytree flattening: dict keys sorted; 'params' < 'x'
    cq = in(idx++);
    auto ff = [&](Ffn& F) { lin(F.w1, 2 * D_FF, D_MODEL); lin(F.w2, D_MODEL, D_FF); };
    ff(ffn_dec); ff(ffn_enc);
    auto lnp = [&](Ln& L) { L.b = in(idx++); L.g = in(idx++); };     // 'b' < 'g'
    lnp(ln_comp); lnp(ln_dec_ffn); lnp(ln_dec_mha); lnp(ln_decomp); lnp(ln_enc_ffn); lnp(ln_enc_mha);
    auto mh = [&](Mha& M) { lin(M.k,512,512); lin(M.o,512,512); lin(M.q,512,512); lin(M.v,512,512); };
    mh(mha_comp); mh(mha_dec); mh(mha_decomp); mh(mha_enc);
    x = in(idx++);
  }

  // ---- workspace bump allocator ----
  char* wp = (char*)d_ws;
  auto alloc = [&](size_t bytes) -> void* {
    void* r = (void*)wp;
    wp += (bytes + 255) & ~(size_t)255;
    return r;
  };

  auto cvt = [&](const float* src, size_t n) -> unsigned short* {
    unsigned short* d = (unsigned short*)alloc(n * 2);
    cvt_bf16_kernel<<<(unsigned)((n + 255) / 256), 256, 0, stream>>>(src, d, n);
    return d;
  };
  auto prep_lin = [&](LinP& L) { L.Wb = cvt(L.W, (size_t)L.n * L.k); };
  auto prep_mha = [&](Mha& M) { prep_lin(M.q); prep_lin(M.k); prep_lin(M.v); prep_lin(M.o); };
  prep_mha(mha_enc); prep_mha(mha_dec); prep_mha(mha_comp); prep_mha(mha_decomp);
  prep_lin(ffn_enc.w1); prep_lin(ffn_enc.w2); prep_lin(ffn_dec.w1); prep_lin(ffn_dec.w2);

  unsigned short* cqb = (unsigned short*)alloc((size_t)64 * D_MODEL * 2);
  padcq_kernel<<<128, 256, 0, stream>>>(cq, cqb);

  // ---- transient buffers (sized for the largest level: 16384 tokens) ----
  const int NTMAX = 16384;
  unsigned short* nb  = (unsigned short*)alloc((size_t)NTMAX * D_MODEL * 2);
  float* qf           = (float*)alloc((size_t)NTMAX * D_MODEL * 4);
  float* kf           = (float*)alloc((size_t)NTMAX * D_MODEL * 4);
  float* vf           = (float*)alloc((size_t)NTMAX * D_MODEL * 4);
  unsigned short* Ob  = (unsigned short*)alloc((size_t)NTMAX * D_MODEL * 2);
  float* hh           = (float*)alloc((size_t)NTMAX * (2 * D_FF) * 4);
  unsigned short* glu = (unsigned short*)alloc((size_t)NTMAX * D_FF * 2);

  // ---- per-level persistent buffers ----
  float* h0 = (float*)alloc((size_t)16384 * D_MODEL * 4);
  float* h1 = (float*)alloc((size_t)2048  * D_MODEL * 4);
  float* h2 = (float*)alloc((size_t)16384 * D_MODEL * 4);
  unsigned short* nbc0 = (unsigned short*)alloc((size_t)16384 * D_MODEL * 2);
  unsigned short* nbc1 = (unsigned short*)alloc((size_t)2048  * D_MODEL * 2);
  unsigned short* nbc2 = (unsigned short*)alloc((size_t)16384 * D_MODEL * 2);
  float* comp0  = (float*)alloc((size_t)2048 * D_MODEL * 4);
  float* comp1  = (float*)alloc((size_t)256  * D_MODEL * 4);
  float* x1     = (float*)alloc((size_t)2048 * D_MODEL * 4);
  float* x2     = (float*)alloc((size_t)256  * D_MODEL * 4);
  float* y2     = (float*)alloc((size_t)256  * D_MODEL * 4);
  float* cr1    = (float*)alloc((size_t)256  * D_MODEL * 4);
  float* cr0    = (float*)alloc((size_t)2048 * D_MODEL * 4);
  float* outer0 = (float*)alloc((size_t)2048 * D_MODEL * 4);
  float* outer1 = (float*)alloc((size_t)256  * D_MODEL * 4);
  float* outer2 = (float*)alloc((size_t)2048 * D_MODEL * 4);

  // ---- launch helpers ----
  auto gemm = [&](const unsigned short* A, const LinP& L, const float* res, float* C, int M) {
    dim3 g(L.n / 64, M / 64);
    gemm_bf16<<<g, 256, 0, stream>>>(A, L.Wb, L.b, res, C, M, L.n, L.k);
  };
  auto lnorm = [&](const float* src, const Ln& L, unsigned short* dst, int ntok) {
    ln_kernel<<<ntok / 8, 256, 0, stream>>>(src, L.g, L.b, dst, ntok);
  };
  auto self_block = [&](float* h, int ntok, int nch,
                        const Ln& lnm, const Mha& M, const Ln& lnf, const Ffn& F) {
    lnorm(h, lnm, nb, ntok);
    gemm(nb, M.q, nullptr, qf, ntok);
    gemm(nb, M.k, nullptr, kf, ntok);
    gemm(nb, M.v, nullptr, vf, ntok);
    attn_kernel<<<dim3(nch, NHEADS), 256, 0, stream>>>(qf, kf, vf, Ob, 64, 64, 64, 1, 0);
    gemm(Ob, M.o, h, h, ntok);                                // + residual
    lnorm(h, lnf, nb, ntok);
    gemm(nb, F.w1, nullptr, hh, ntok);
    const size_t n = (size_t)ntok * D_FF;
    siglu_kernel<<<(unsigned)((n + 255) / 256), 256, 0, stream>>>(hh, glu, n);
    gemm(glu, F.w2, h, h, ntok);                              // + residual
  };
  auto compress = [&](float* h, int ntok, int nch, unsigned short* nbc, float* compOut) {
    lnorm(h, ln_comp, nbc, ntok);                             // persists for decompress-q
    gemm(cqb, mha_comp.q, nullptr, qf, 64);                   // 8 queries padded to 64
    gemm(nbc, mha_comp.k, nullptr, kf, ntok);
    gemm(nbc, mha_comp.v, nullptr, vf, ntok);
    attn_kernel<<<dim3(nch, NHEADS), 256, 0, stream>>>(qf, kf, vf, Ob, COMP, 64, 64, 0, 1);
    gemm(Ob, mha_comp.o, nullptr, compOut, nch * COMP);
  };
  auto decompress = [&](float* h, int ntok, int nch, const unsigned short* nbc, const float* outer) {
    lnorm(outer, ln_decomp, nb, nch * COMP);
    gemm(nbc, mha_decomp.q, nullptr, qf, ntok);
    gemm(nb,  mha_decomp.k, nullptr, kf, nch * COMP);
    gemm(nb,  mha_decomp.v, nullptr, vf, nch * COMP);
    attn_kernel<<<dim3(nch, NHEADS), 256, 0, stream>>>(qf, kf, vf, Ob, 64, COMP, 32, 0, 0);
    gemm(Ob, mha_decomp.o, h, h, ntok);                       // + pre residual
  };
  auto gs = [](size_t n) { return (unsigned)((n + 255) / 256); };

  // =========================== level 0 (part A) ============================
  const size_t n0 = (size_t)16384 * D_MODEL;
  copy_f32_kernel<<<gs(n0), 256, 0, stream>>>(x, h0, n0);
  self_block(h0, 16384, 256, ln_enc_mha, mha_enc, ln_enc_ffn, ffn_enc);
  compress(h0, 16384, 256, nbc0, comp0);
  permute_kernel<<<gs((size_t)2048 * D_MODEL), 256, 0, stream>>>(comp0, x1, 4, 64, 0);

  // =========================== level 1 (part A) ============================
  const size_t n1 = (size_t)2048 * D_MODEL;
  copy_f32_kernel<<<gs(n1), 256, 0, stream>>>(x1, h1, n1);
  self_block(h1, 2048, 32, ln_enc_mha, mha_enc, ln_enc_ffn, ffn_enc);
  compress(h1, 2048, 32, nbc1, comp1);
  permute_kernel<<<gs((size_t)256 * D_MODEL), 256, 0, stream>>>(comp1, x2, 32, 1, 0);

  // ======================= level 2 (base case, full) =======================
  pad_seq1_kernel<<<gs(n0), 256, 0, stream>>>(x2, h2, 256);
  self_block(h2, 16384, 256, ln_enc_mha, mha_enc, ln_enc_ffn, ffn_enc);
  lnorm(h2, ln_comp, nbc2, 16384);                            // normed (no compress: nf==0)
  build_outer_kernel<<<gs((size_t)2048 * D_MODEL), 256, 0, stream>>>(nullptr, cq, outer2, 256, 1);
  decompress(h2, 16384, 256, nbc2, outer2);
  self_block(h2, 16384, 256, ln_dec_mha, mha_dec, ln_dec_ffn, ffn_dec);
  extract_seq1_kernel<<<gs((size_t)256 * D_MODEL), 256, 0, stream>>>(h2, y2, 256);

  // =========================== level 1 (part B) ============================
  permute_kernel<<<gs((size_t)256 * D_MODEL), 256, 0, stream>>>(y2, cr1, 32, 1, 1);
  build_outer_kernel<<<gs((size_t)256 * D_MODEL), 256, 0, stream>>>(cr1, cq, outer1, 32, 1);
  decompress(h1, 2048, 32, nbc1, outer1);
  self_block(h1, 2048, 32, ln_dec_mha, mha_dec, ln_dec_ffn, ffn_dec);

  // =========================== level 0 (part B) ============================
  permute_kernel<<<gs((size_t)2048 * D_MODEL), 256, 0, stream>>>(h1, cr0, 4, 64, 1);
  build_outer_kernel<<<gs((size_t)2048 * D_MODEL), 256, 0, stream>>>(cr0, cq, outer0, 4, 64);
  decompress(h0, 16384, 256, nbc0, outer0);
  self_block(h0, 16384, 256, ln_dec_mha, mha_dec, ln_dec_ffn, ffn_dec);

  copy_f32_kernel<<<gs(n0), 256, 0, stream>>>(h0, (float*)d_out, n0);
}
